// GAT_Net_12455405159160
// MI455X (gfx1250) — matrix operand order repaired and checked by reading, not compiled, not run on hardware
//
#include <hip/hip_runtime.h>
#include <hip/hip_bf16.h>
#include <stdint.h>

// ---------------------------------------------------------------------------
// GAT 2-layer inference for MI455X (gfx1250, wave32, WMMA).
//   N=100000 nodes, E=1.6M edges (+N self loops), F_IN=128,
//   layer1: 4 heads x 32 ch (=128), layer2: 1 head x 16 ch.
// ---------------------------------------------------------------------------

typedef __attribute__((ext_vector_type(16))) _Float16 v16h;
typedef __attribute__((ext_vector_type(8)))  float    v8f;

#define F_IN      128
#define NEG_SLOPE 0.2f

// ---------------------------------------------------------------------------
// WMMA GEMM: OUT[nrows, NT*16] = X[nrows,128] * W[NT*16,128]^T  (f16 mul, f32 acc)
// Block = 256 threads = 8 waves; wave w owns row-tile (blockIdx.x*8 + w).
// W is staged into LDS once per block, pre-swizzled into B-fragment layout:
//   fragment(kstep, nt), lane L holds column n = nt*16 + (L&15),
//   K = kstep*32 + (L>=16 ? 16 : 0) + {0..15}, packed 2 f16 per dword.
// A-fragment matches B: lane<16 -> K {0..15},
// lane>=16 -> K {16..31} of the kstep's 32-wide K slice.
// ---------------------------------------------------------------------------
template<int NT>
__global__ void wmma_gemm_f16(const float* __restrict__ X,
                              const float* __restrict__ W,
                              float* __restrict__ OUT,
                              int nrows)
{
    __shared__ uint32_t bfrag[NT * 4 * 32 * 8];   // NT*4 fragments * 32 lanes * 8 dwords

    const int tid = threadIdx.x;

    // ---- stage W (f32 global) -> LDS as packed-f16 B fragments ----
    for (int i = tid; i < NT * 4 * 32 * 8; i += 256) {
        const int d     = i & 7;
        const int lane  = (i >> 3) & 31;
        const int f     = i >> 8;          // kstep*NT + nt
        const int kstep = f / NT;
        const int nt    = f % NT;
        const int ncol  = nt * 16 + (lane & 15);
        const int K     = kstep * 32 + ((lane >> 4) * 16) + 2 * d;
        union { _Float16 h[2]; uint32_t u; } pk;
        pk.h[0] = (_Float16)W[(size_t)ncol * F_IN + K];
        pk.h[1] = (_Float16)W[(size_t)ncol * F_IN + K + 1];
        bfrag[i] = pk.u;
    }
    __syncthreads();

    const int wave     = tid >> 5;
    const int lane     = tid & 31;
    const int rt       = blockIdx.x * 8 + wave;
    const int row_base = rt * 16;
    const int m_lane   = row_base + (lane & 15);

    v8f acc[NT];
    const v8f vzero = {};
    #pragma unroll
    for (int nt = 0; nt < NT; ++nt) acc[nt] = vzero;

    for (int kstep = 0; kstep < 4; ++kstep) {
        // ---- A fragment: 16 f16 per lane from X (fp32 -> f16 convert) ----
        union { _Float16 h[16]; v16h v; } af;
        const int kbase = kstep * 32 + ((lane >= 16) ? 16 : 0);
        if (m_lane < nrows) {
            const float* xr = X + (size_t)m_lane * F_IN + kbase;
            #pragma unroll
            for (int j = 0; j < 8; ++j) af.h[j]     = (_Float16)xr[j];
            #pragma unroll
            for (int j = 0; j < 8; ++j) af.h[8 + j] = (_Float16)xr[8 + j];
        } else {
            #pragma unroll
            for (int j = 0; j < 16; ++j) af.h[j] = (_Float16)0.0f;
        }

        #pragma unroll
        for (int nt = 0; nt < NT; ++nt) {
            union { uint32_t u[8]; v16h v; } bf;
            const uint32_t* p = &bfrag[(size_t)((kstep * NT + nt) * 32 + lane) * 8];
            #pragma unroll
            for (int d = 0; d < 8; ++d) bf.u[d] = p[d];
            acc[nt] = __builtin_amdgcn_wmma_f32_16x16x32_f16(
                false, af.v, false, bf.v, (short)0, acc[nt], false, false);
        }
    }

    // ---- store: C/D layout — lane<16: VGPR j -> M=j; lane>=16: M=8+j; N=lane&15
    #pragma unroll
    for (int nt = 0; nt < NT; ++nt) {
        const int col = nt * 16 + (lane & 15);
        #pragma unroll
        for (int j = 0; j < 8; ++j) {
            const int m = row_base + ((lane < 16) ? j : (8 + j));
            if (m < nrows) OUT[(size_t)m * (NT * 16) + col] = acc[nt][j];
        }
    }
}

// ---------------------------------------------------------------------------
// Per-node attention half scores: al[n,h] = sum_c H[n, h*cph + c] * a[h*cph + c]
// ---------------------------------------------------------------------------
__global__ void att_scores(const float* __restrict__ H,
                           const float* __restrict__ a_src,
                           const float* __restrict__ a_dst,
                           float* __restrict__ alS,
                           float* __restrict__ alD,
                           int n, int heads, int cph)
{
    const int idx = blockIdx.x * blockDim.x + threadIdx.x;
    if (idx >= n * heads) return;
    const int node = idx / heads;
    const int h    = idx % heads;
    const float* row = H + (size_t)node * (heads * cph) + h * cph;
    const float* as  = a_src + h * cph;
    const float* ad  = a_dst + h * cph;
    float s = 0.f, d = 0.f;
    for (int c = 0; c < cph; ++c) {
        const float v = row[c];
        s += v * as[c];
        d += v * ad[c];
    }
    alS[idx] = s;
    alD[idx] = d;
}

// ---------------------------------------------------------------------------
// float atomic max via sign-split int trick (init value must be -inf)
// ---------------------------------------------------------------------------
__device__ __forceinline__ void atomicMaxF(float* addr, float v)
{
    if (v >= 0.f) atomicMax((int*)addr,          __float_as_int(v));
    else          atomicMin((unsigned int*)addr, __float_as_uint(v));
}

// Pass A: segment max of leaky-relu edge scores into m[dst,h].
// Self-loops are synthesized for e >= E.
__global__ void edge_max(const int* __restrict__ ei, int E, int n, int heads,
                         const float* __restrict__ alS,
                         const float* __restrict__ alD,
                         float* __restrict__ m)
{
    const int idx = blockIdx.x * blockDim.x + threadIdx.x;
    if (idx >= (E + n) * heads) return;
    const int e = idx / heads;
    const int h = idx % heads;
    const int src = (e < E) ? ei[e]     : (e - E);
    const int dst = (e < E) ? ei[E + e] : (e - E);
    float sc = alS[src * heads + h] + alD[dst * heads + h];
    sc = (sc >= 0.f) ? sc : NEG_SLOPE * sc;
    atomicMaxF(&m[dst * heads + h], sc);
}

// Pass B: w = exp(score - m[dst]); z[dst,h] += w; acc[dst, c] += w * H[src, c].
// One thread per (edge, 4 channels): coalesced float4 gather + 4 f32 atomics.
__global__ void edge_aggregate(const int* __restrict__ ei, int E, int n,
                               int CH, int heads,
                               const float* __restrict__ alS,
                               const float* __restrict__ alD,
                               const float* __restrict__ m,
                               const float* __restrict__ H,
                               float* __restrict__ acc,
                               float* __restrict__ z)
{
    const int       tpe   = CH >> 2;                  // threads per edge (VEC=4)
    const int       cph   = CH / heads;
    const long long tid   = (long long)blockIdx.x * blockDim.x + threadIdx.x;
    const long long total = (long long)(E + n) * tpe;
    if (tid >= total) return;

    const int e  = (int)(tid / tpe);
    const int c0 = (int)(tid % tpe) << 2;
    const int h  = c0 / cph;
    const int src = (e < E) ? ei[e]     : (e - E);
    const int dst = (e < E) ? ei[E + e] : (e - E);

    float sc = alS[src * heads + h] + alD[dst * heads + h];
    sc = (sc >= 0.f) ? sc : NEG_SLOPE * sc;
    const float w = __expf(sc - m[dst * heads + h]);

    if ((c0 % cph) == 0) atomicAdd(&z[dst * heads + h], w);

    const float4 hv = *(const float4*)(H + (size_t)src * CH + c0);
    float* ap = acc + (size_t)dst * CH + c0;
    atomicAdd(ap + 0, w * hv.x);
    atomicAdd(ap + 1, w * hv.y);
    atomicAdd(ap + 2, w * hv.z);
    atomicAdd(ap + 3, w * hv.w);
}

// Pass C: out = acc / (z + 1e-16) + bias, optional relu.
__global__ void finalize(const float* __restrict__ acc,
                         const float* __restrict__ z,
                         const float* __restrict__ bias,
                         float* __restrict__ out,
                         int n, int ch, int heads, int do_relu)
{
    const int idx = blockIdx.x * blockDim.x + threadIdx.x;
    if (idx >= n * ch) return;
    const int node = idx / ch;
    const int c    = idx % ch;
    const int h    = c / (ch / heads);
    float v = acc[idx] / (z[node * heads + h] + 1e-16f) + bias[c];
    if (do_relu) v = fmaxf(v, 0.f);
    out[idx] = v;
}

__global__ void fill_f32(float* __restrict__ p, float v, long long count)
{
    const long long idx = (long long)blockIdx.x * blockDim.x + threadIdx.x;
    if (idx < count) p[idx] = v;
}

static inline int blocks_for(long long total, int bs) {
    return (int)((total + bs - 1) / bs);
}

extern "C" void kernel_launch(void* const* d_in, const int* in_sizes, int n_in,
                              void* d_out, int out_size, void* d_ws, size_t ws_size,
                              hipStream_t stream)
{
    (void)n_in; (void)out_size; (void)ws_size;

    const float* x   = (const float*)d_in[0];
    const int*   ei  = (const int*)  d_in[1];
    const float* W1  = (const float*)d_in[2];
    const float* aS1 = (const float*)d_in[3];
    const float* aD1 = (const float*)d_in[4];
    const float* b1  = (const float*)d_in[5];
    const float* W2  = (const float*)d_in[6];
    const float* aS2 = (const float*)d_in[7];
    const float* aD2 = (const float*)d_in[8];
    const float* b2  = (const float*)d_in[9];
    float*       out = (float*)d_out;

    const int n = in_sizes[0] / F_IN;   // 100000
    const int E = in_sizes[1] / 2;      // 1.6M
    const int H1 = 4, C1 = 128;         // layer1: 4 heads * 32 ch
    const int H2 = 1, C2 = 16;          // layer2: 1 head  * 16 ch

    // ---- workspace carve-up (floats) ----
    float* ws = (float*)d_ws;
    size_t o = 0;
    float* h1   = ws + o; o += (size_t)n * C1;   // 12.8M
    float* acc1 = ws + o; o += (size_t)n * C1;   // 12.8M
    float* h2   = ws + o; o += (size_t)n * C2;   // 1.6M
    float* acc2 = ws + o; o += (size_t)n * C2;   // 1.6M
    float* alS1v = ws + o; o += (size_t)n * H1;
    float* alD1v = ws + o; o += (size_t)n * H1;
    float* m1    = ws + o; o += (size_t)n * H1;
    float* z1    = ws + o; o += (size_t)n * H1;
    float* alS2v = ws + o; o += (size_t)n * H2;
    float* alD2v = ws + o; o += (size_t)n * H2;
    float* m2    = ws + o; o += (size_t)n * H2;
    float* z2    = ws + o; o += (size_t)n * H2;

    const float NEG_INF = -__builtin_huge_valf();
    const int BS = 256;

    // ---- re-init accumulators every call (deterministic across graph replays)
    fill_f32<<<blocks_for((long long)n * C1, BS), BS, 0, stream>>>(acc1, 0.f, (long long)n * C1);
    fill_f32<<<blocks_for((long long)n * C2, BS), BS, 0, stream>>>(acc2, 0.f, (long long)n * C2);
    fill_f32<<<blocks_for((long long)n * H1, BS), BS, 0, stream>>>(m1, NEG_INF, (long long)n * H1);
    fill_f32<<<blocks_for((long long)n * H1, BS), BS, 0, stream>>>(z1, 0.f,     (long long)n * H1);
    fill_f32<<<blocks_for((long long)n * H2, BS), BS, 0, stream>>>(m2, NEG_INF, (long long)n * H2);
    fill_f32<<<blocks_for((long long)n * H2, BS), BS, 0, stream>>>(z2, 0.f,     (long long)n * H2);

    // ---- layer 1 ----
    const int gemm_blocks = (n + 127) / 128;   // 8 waves/block * 16 rows/wave
    wmma_gemm_f16<8><<<gemm_blocks, 256, 0, stream>>>(x, W1, h1, n);

    att_scores<<<blocks_for((long long)n * H1, BS), BS, 0, stream>>>(
        h1, aS1, aD1, alS1v, alD1v, n, H1, C1 / H1);

    edge_max<<<blocks_for((long long)(E + n) * H1, BS), BS, 0, stream>>>(
        ei, E, n, H1, alS1v, alD1v, m1);

    edge_aggregate<<<blocks_for((long long)(E + n) * (C1 / 4), BS), BS, 0, stream>>>(
        ei, E, n, C1, H1, alS1v, alD1v, m1, h1, acc1, z1);

    finalize<<<blocks_for((long long)n * C1, BS), BS, 0, stream>>>(
        acc1, z1, b1, acc1, n, C1, H1, /*relu=*/1);

    // ---- layer 2 ----
    wmma_gemm_f16<1><<<gemm_blocks, 256, 0, stream>>>(acc1, W2, h2, n);

    att_scores<<<blocks_for((long long)n * H2, BS), BS, 0, stream>>>(
        h2, aS2, aD2, alS2v, alD2v, n, H2, C2 / H2);

    edge_max<<<blocks_for((long long)(E + n) * H2, BS), BS, 0, stream>>>(
        ei, E, n, H2, alS2v, alD2v, m2);

    edge_aggregate<<<blocks_for((long long)(E + n) * (C2 / 4), BS), BS, 0, stream>>>(
        ei, E, n, C2, H2, alS2v, alD2v, m2, h2, acc2, z2);

    finalize<<<blocks_for((long long)n * C2, BS), BS, 0, stream>>>(
        acc2, z2, b2, out, n, C2, H2, /*relu=*/0);
}
